// MoeHashV2Layer_40853728919572
// MI455X (gfx1250) — matrix operand order repaired and checked by reading, not compiled, not run on hardware
//
#include <hip/hip_runtime.h>
#include <hip/hip_bf16.h>

#define DIM  768
#define HID  3072
#define NE   8
#define NTOK 4096

typedef __bf16 bf16_t;
typedef __attribute__((ext_vector_type(16))) __bf16 v16bf;
typedef __attribute__((ext_vector_type(8)))  float  v8f;

struct __align__(16) U4 { unsigned x, y, z, w; };

union Frag {
    v16bf v;
    U4    q[2];
};

__device__ __forceinline__ bf16_t f2bf(float f) {
    union { float f; unsigned u; } a; a.f = f;
    unsigned r = a.u + 0x7FFFu + ((a.u >> 16) & 1u);   // round-to-nearest-even
    union { unsigned short s; bf16_t b; } o;
    o.s = (unsigned short)(r >> 16);
    return o.b;
}

// ---- CDNA5 async global->LDS copy (16B per lane), tracked by ASYNCcnt ----
__device__ __forceinline__ void async_ld16(unsigned ldsAddr, const void* g) {
    asm volatile("global_load_async_to_lds_b128 %0, %1, off"
                 :: "v"(ldsAddr), "v"(g)
                 : "memory");
}
__device__ __forceinline__ void wait_async0() {
    asm volatile("s_wait_asynccnt 0x0" ::: "memory");
}
__device__ __forceinline__ unsigned lds_off(const void* p) {
    return (unsigned)(size_t)p;   // low 32 bits of flat shared address = LDS offset
}

// ---------------- bucketing ----------------
__global__ void k_zero(int* cnt) {
    if (threadIdx.x < NE) cnt[threadIdx.x] = 0;
}

__global__ void k_bucket(const int* __restrict__ mt, int* cnt, int* idx) {
    int n = blockIdx.x * 256 + threadIdx.x;
    if (n < NTOK) {
        int e = mt[n];
        int p = atomicAdd(&cnt[e], 1);
        idx[e * NTOK + p] = n;
    }
}

// ---------------- f32 -> bf16 convert of x ----------------
__global__ __launch_bounds__(256) void k_cvt_x(const float* __restrict__ x,
                                               bf16_t* __restrict__ xb) {
    int i = (blockIdx.x * 256 + threadIdx.x) * 4;
#pragma unroll
    for (int j = 0; j < 4; ++j) xb[i + j] = f2bf(x[i + j]);
}

// ---------------- tiled transpose: in [R][C] f32 -> out [C][R] bf16, per expert ----------------
__global__ __launch_bounds__(256) void k_transpose(const float* __restrict__ in,
                                                   bf16_t* __restrict__ out,
                                                   int R, int C) {
    __shared__ float tile[32][33];
    const float* inp  = in  + (size_t)blockIdx.z * R * C;
    bf16_t*      outp = out + (size_t)blockIdx.z * R * C;
    int tx = threadIdx.x, ty = threadIdx.y;         // blockDim = (32, 8)
    int c = blockIdx.x * 32 + tx;
    int rbase = blockIdx.y * 32;
#pragma unroll
    for (int i = 0; i < 4; ++i) {
        int r = rbase + ty + i * 8;
        tile[ty + i * 8][tx] = inp[(size_t)r * C + c];
    }
    __syncthreads();
    int oc = rbase + tx;                             // out col = original row
    int orbase = blockIdx.x * 32;                    // out row = original col
#pragma unroll
    for (int i = 0; i < 4; ++i) {
        int orr = orbase + ty + i * 8;
        outp[(size_t)orr * R + oc] = f2bf(tile[tx][ty + i * 8]);
    }
}

// =====================================================================
// LDS staging swizzle: tiles stored in the exact WMMA fragment register
// image, so compute reads are two ds_load_b128 per fragment.
//   A (16x32 bf16): lane = (kchunk&1)*16 + row%16, half = kchunk>>1
//   B (32x16 bf16): lane = (kchunk&1)*16 + col%16, half = kchunk>>1
// Each 64(row/col) x 32(k) bf16 tile = 256 U4 chunks = 4 KB.
// Staging uses GLOBAL_LOAD_ASYNC_TO_LDS_B128 (ASYNCcnt), double-buffered.
// =====================================================================

// ---------------- gate/up GEMM + SiLU*up fused epilogue ----------------
// grid: (HID/64, NTOK/64, NE), block 256 (8 waves). Tile: M=64, N=64.
__global__ __launch_bounds__(256) void k_gateup(
    const bf16_t* __restrict__ xb,  const bf16_t* __restrict__ wgt,
    const bf16_t* __restrict__ wit, const float*  __restrict__ bg,
    const float*  __restrict__ bi,  const int*    __restrict__ idx,
    const int*    __restrict__ cnt, bf16_t*       __restrict__ hbuf)
{
    __shared__ U4 sA[2][256];
    __shared__ U4 sG[2][256];
    __shared__ U4 sU[2][256];

    const int e  = blockIdx.z;
    const int m0 = blockIdx.y * 64;
    const int n0 = blockIdx.x * 64;
    const int ce = cnt[e];
    if (m0 >= ce) return;                 // uniform across block

    const int tid   = threadIdx.x;
    const int wave  = tid >> 5;
    const int lane  = tid & 31;
    const int lsub  = lane & 15;
    const int lhalf = lane >> 4;

    const int mt  = wave & 3;             // m-tile 0..3
    const int nt0 = (wave >> 2) * 2;      // n-tiles {nt0, nt0+1}

    // ---- staging setup: thread -> (row/col sr, k-chunk sc) ----
    const int sr = tid >> 2;              // 0..63
    const int sc = tid & 3;               // 0..3 (8 bf16 each)
    int strc = (m0 + sr) < ce ? (m0 + sr) : (ce - 1);
    int stok = idx[e * NTOK + strc];
    const size_t ew = (size_t)e * HID * DIM;
    const bf16_t* gA = xb  + (size_t)stok * DIM + sc * 8;
    const bf16_t* gG = wgt + ew + (size_t)(n0 + sr) * DIM + sc * 8;
    const bf16_t* gU = wit + ew + (size_t)(n0 + sr) * DIM + sc * 8;
    const int aDst = ((sr >> 4) << 6) + (((((sc & 1) << 4) | (sr & 15))) << 1) + (sc >> 1);
    const int bDst = ((sr >> 4) << 6) + (((((sc & 1) << 4) | (sr & 15))) << 1) + (sc >> 1);
    const unsigned lA0 = lds_off(&sA[0][aDst]), lA1 = lds_off(&sA[1][aDst]);
    const unsigned lG0 = lds_off(&sG[0][bDst]), lG1 = lds_off(&sG[1][bDst]);
    const unsigned lU0 = lds_off(&sU[0][bDst]), lU1 = lds_off(&sU[1][bDst]);

    // ---- compute-side LDS offsets ----
    const int aOff  = (mt << 6) + (lane << 1);
    const int b0Off = (nt0 << 6) + (lane << 1);
    const int b1Off = ((nt0 + 1) << 6) + (lane << 1);

    // preload stage 0 (async)
    async_ld16(lA0, gA);
    async_ld16(lG0, gG);
    async_ld16(lU0, gU);
    wait_async0();
    __syncthreads();

    v8f ag0 = {}, ag1 = {}, au0 = {}, au1 = {};

    for (int kk = 0; kk < DIM; kk += 32) {
        const int s = (kk >> 5) & 1;
        if (kk + 32 < DIM) {              // async-stage next k-slice during compute
            async_ld16(s ? lA0 : lA1, gA + kk + 32);
            async_ld16(s ? lG0 : lG1, gG + kk + 32);
            async_ld16(s ? lU0 : lU1, gU + kk + 32);
        }
        Frag a, g0, g1, u0, u1;
        a.q[0]  = sA[s][aOff];      a.q[1]  = sA[s][aOff + 1];
        g0.q[0] = sG[s][b0Off];     g0.q[1] = sG[s][b0Off + 1];
        g1.q[0] = sG[s][b1Off];     g1.q[1] = sG[s][b1Off + 1];
        u0.q[0] = sU[s][b0Off];     u0.q[1] = sU[s][b0Off + 1];
        u1.q[0] = sU[s][b1Off];     u1.q[1] = sU[s][b1Off + 1];

        ag0 = __builtin_amdgcn_wmma_f32_16x16x32_bf16(false, a.v, false, g0.v, (short)0, ag0, false, false);
        ag1 = __builtin_amdgcn_wmma_f32_16x16x32_bf16(false, a.v, false, g1.v, (short)0, ag1, false, false);
        au0 = __builtin_amdgcn_wmma_f32_16x16x32_bf16(false, a.v, false, u0.v, (short)0, au0, false, false);
        au1 = __builtin_amdgcn_wmma_f32_16x16x32_bf16(false, a.v, false, u1.v, (short)0, au1, false, false);

        wait_async0();                    // next stage landed in LDS
        __syncthreads();
    }

#pragma unroll
    for (int j = 0; j < 2; ++j) {
        const v8f ag = j ? ag1 : ag0;
        const v8f au = j ? au1 : au0;
        int ncol = n0 + (nt0 + j) * 16 + lsub;
        float bgv = bg[(size_t)e * HID + ncol];
        float biv = bi[(size_t)e * HID + ncol];
#pragma unroll
        for (int r = 0; r < 8; ++r) {
            int row = m0 + mt * 16 + r + 8 * lhalf;
            if (row < ce) {
                int t = idx[e * NTOK + row];
                float gv = ag[r] + bgv;
                float uv = au[r] + biv;
                float hv = (gv / (1.0f + __expf(-gv))) * uv;   // silu(g) * u
                hbuf[(size_t)t * HID + ncol] = f2bf(hv);
            }
        }
    }
}

// ---------------- down GEMM + bias, scatter f32 output ----------------
// grid: (DIM/64, NTOK/64, NE), block 256. Tile: M=64, N=64, K=3072.
__global__ __launch_bounds__(256) void k_down(
    const bf16_t* __restrict__ hbuf, const bf16_t* __restrict__ wot,
    const float*  __restrict__ bo,   const int*    __restrict__ idx,
    const int*    __restrict__ cnt,  float*        __restrict__ out)
{
    __shared__ U4 sA[2][256];
    __shared__ U4 sB[2][256];

    const int e  = blockIdx.z;
    const int m0 = blockIdx.y * 64;
    const int n0 = blockIdx.x * 64;
    const int ce = cnt[e];
    if (m0 >= ce) return;

    const int tid   = threadIdx.x;
    const int wave  = tid >> 5;
    const int lane  = tid & 31;
    const int lsub  = lane & 15;
    const int lhalf = lane >> 4;

    const int mt  = wave & 3;
    const int nt0 = (wave >> 2) * 2;

    const int sr = tid >> 2;
    const int sc = tid & 3;
    int strc = (m0 + sr) < ce ? (m0 + sr) : (ce - 1);
    int stok = idx[e * NTOK + strc];
    const size_t ew = (size_t)e * DIM * HID;
    const bf16_t* gA = hbuf + (size_t)stok * HID + sc * 8;
    const bf16_t* gB = wot + ew + (size_t)(n0 + sr) * HID + sc * 8;
    const int aDst = ((sr >> 4) << 6) + (((((sc & 1) << 4) | (sr & 15))) << 1) + (sc >> 1);
    const int bDst = ((sr >> 4) << 6) + (((((sc & 1) << 4) | (sr & 15))) << 1) + (sc >> 1);
    const unsigned lA0 = lds_off(&sA[0][aDst]), lA1 = lds_off(&sA[1][aDst]);
    const unsigned lB0 = lds_off(&sB[0][bDst]), lB1 = lds_off(&sB[1][bDst]);

    const int aOff  = (mt << 6) + (lane << 1);
    const int b0Off = (nt0 << 6) + (lane << 1);
    const int b1Off = ((nt0 + 1) << 6) + (lane << 1);

    async_ld16(lA0, gA);
    async_ld16(lB0, gB);
    wait_async0();
    __syncthreads();

    v8f c0 = {}, c1 = {};

    for (int kk = 0; kk < HID; kk += 32) {
        const int s = (kk >> 5) & 1;
        if (kk + 32 < HID) {
            async_ld16(s ? lA0 : lA1, gA + kk + 32);
            async_ld16(s ? lB0 : lB1, gB + kk + 32);
        }
        Frag a, b0, b1;
        a.q[0]  = sA[s][aOff];   a.q[1]  = sA[s][aOff + 1];
        b0.q[0] = sB[s][b0Off];  b0.q[1] = sB[s][b0Off + 1];
        b1.q[0] = sB[s][b1Off];  b1.q[1] = sB[s][b1Off + 1];

        c0 = __builtin_amdgcn_wmma_f32_16x16x32_bf16(false, a.v, false, b0.v, (short)0, c0, false, false);
        c1 = __builtin_amdgcn_wmma_f32_16x16x32_bf16(false, a.v, false, b1.v, (short)0, c1, false, false);

        wait_async0();
        __syncthreads();
    }

#pragma unroll
    for (int j = 0; j < 2; ++j) {
        const v8f cc = j ? c1 : c0;
        int ncol = n0 + (nt0 + j) * 16 + lsub;
        float bov = bo[(size_t)e * DIM + ncol];
#pragma unroll
        for (int r = 0; r < 8; ++r) {
            int row = m0 + mt * 16 + r + 8 * lhalf;
            if (row < ce) {
                int t = idx[e * NTOK + row];
                out[(size_t)t * DIM + ncol] = cc[r] + bov;
            }
        }
    }
}

// ---------------- host ----------------
extern "C" void kernel_launch(void* const* d_in, const int* in_sizes, int n_in,
                              void* d_out, int out_size, void* d_ws, size_t ws_size,
                              hipStream_t stream) {
    (void)in_sizes; (void)n_in; (void)out_size; (void)ws_size;
    const float* x   = (const float*)d_in[0];
    const int*   map = (const int*)  d_in[1];
    const float* Wg  = (const float*)d_in[2];
    const float* bg  = (const float*)d_in[3];
    const float* Wi  = (const float*)d_in[4];
    const float* bi  = (const float*)d_in[5];
    const float* Wo  = (const float*)d_in[6];
    const float* bo  = (const float*)d_in[7];
    float* out = (float*)d_out;

    char* ws = (char*)d_ws;
    int*    cnt = (int*)(ws + 0);              //    32 B
    int*    idx = (int*)(ws + 256);            //   128 KB
    bf16_t* xb  = (bf16_t*)(ws + 262144);      //  6.29 MB
    bf16_t* wgt = (bf16_t*)(ws + 6553600);     // 37.75 MB  [E][HID][DIM] bf16
    bf16_t* wit = (bf16_t*)(ws + 44302336);    // 37.75 MB  [E][HID][DIM] bf16
    bf16_t* wot = (bf16_t*)(ws + 82051072);    // 37.75 MB  [E][DIM][HID] bf16
    bf16_t* hb  = (bf16_t*)(ws + 119799808);   // 25.17 MB  [NTOK][HID]   bf16

    k_zero  <<<1, 32, 0, stream>>>(cnt);
    k_bucket<<<NTOK / 256, 256, 0, stream>>>(map, cnt, idx);
    k_cvt_x <<<(NTOK * DIM) / 1024, 256, 0, stream>>>(x, xb);
    k_transpose<<<dim3(HID / 32, DIM / 32, NE), dim3(32, 8), 0, stream>>>(Wg, wgt, DIM, HID);
    k_transpose<<<dim3(HID / 32, DIM / 32, NE), dim3(32, 8), 0, stream>>>(Wi, wit, DIM, HID);
    k_transpose<<<dim3(DIM / 32, HID / 32, NE), dim3(32, 8), 0, stream>>>(Wo, wot, HID, DIM);
    k_gateup<<<dim3(HID / 64, NTOK / 64, NE), 256, 0, stream>>>(xb, wgt, wit, bg, bi, idx, cnt, hb);
    k_down  <<<dim3(DIM / 64, NTOK / 64, NE), 256, 0, stream>>>(hb, wot, bo, idx, cnt, out);
}
